// LinearCrossAttention_40827959116314
// MI455X (gfx1250) — compile-verified, matrix operand order checked by reading, not run on hardware
//
#include <hip/hip_runtime.h>
#include <math.h>
#include <stdint.h>

typedef __attribute__((ext_vector_type(16))) _Float16 v16h;
typedef __attribute__((ext_vector_type(8)))  float    v8f;
typedef __attribute__((ext_vector_type(4)))  unsigned int v4u;
typedef __attribute__((ext_vector_type(8)))  int      v8i;
typedef __attribute__((ext_vector_type(4)))  int      v4i;

static constexpr int BATCH = 4;
static constexpr int NTOK  = 4096;
static constexpr int CT    = 960;
static constexpr int MTOT  = BATCH * NTOK;   // 16384 tokens

__device__ __forceinline__ v8f wmma_f16(v16h a, v16h b, v8f c) {
  return __builtin_amdgcn_wmma_f32_16x16x32_f16(
      /*neg_a=*/false, a, /*neg_b=*/false, b,
      /*c_mod=*/(short)0, c, /*reuse_a=*/false, /*reuse_b=*/false);
}

// ---------------------------------------------------------------------------
// TDM: DMA a 2-D tile (tile1 rows x tile0 f32 elements, row stride = stride0
// elements) from global memory into LDS at byte offset lds_off.
// D# packing per CDNA5 ISA ch.8 (group0 128b, group1 256b; groups 2/3 zero
// for a 2-D tensor). data_size=2 (4 bytes), type=2 ("image"), count=1.
// ---------------------------------------------------------------------------
__device__ __forceinline__ void tdm_load_2d(uint64_t gaddr, uint32_t lds_off,
                                            uint32_t dim0, uint32_t dim1,
                                            uint32_t stride0,
                                            uint32_t tile0, uint32_t tile1)
{
  v4u g0;
  g0[0] = 1u;                                        // count=1 (valid), user D#
  g0[1] = lds_off;                                   // lds_addr (bytes)
  g0[2] = (uint32_t)(gaddr & 0xFFFFFFFFu);           // global_addr[31:0]
  g0[3] = (uint32_t)((gaddr >> 32) & 0x1FFFFFFu)     // global_addr[56:32]
        | (2u << 30);                                // type=2
  v8i g1;
  g1[0] = (int)(2u << 16);                           // data_size=4B
  g1[1] = (int)((dim0 & 0xFFFFu) << 16);             // tensor_dim0 lo16
  g1[2] = (int)(((dim0 >> 16) & 0xFFFFu) | ((dim1 & 0xFFFFu) << 16));
  g1[3] = (int)(((dim1 >> 16) & 0xFFFFu) | ((tile0 & 0xFFFFu) << 16));
  g1[4] = (int)(tile1 & 0xFFFFu);                    // tile_dim1 (tile_dim2=0)
  g1[5] = (int)stride0;                              // tensor_dim0_stride lo32
  g1[6] = 0;                                         // stride0 hi / stride1 lo
  g1[7] = 0;
  v4i z4 = {};
#if defined(__clang_major__) && (__clang_major__ >= 23)
  v8i z8 = {};
  __builtin_amdgcn_tensor_load_to_lds(g0, g1, z4, z4, z8, 0);
#else
  __builtin_amdgcn_tensor_load_to_lds(g0, g1, z4, z4, 0);
#endif
}

// ---------------------------------------------------------------------------
// 1) q[b,n] = dot(x_total[b,n,:], q_w) + q_b   (one wave per token)
// ---------------------------------------------------------------------------
__global__ __launch_bounds__(256) void q_kernel(
    const float* __restrict__ xt, const float* __restrict__ qw,
    const float* __restrict__ qb, float* __restrict__ q)
{
  int wave = (int)((blockIdx.x * blockDim.x + threadIdx.x) >> 5);
  int lane = threadIdx.x & 31;
  if (wave >= MTOT) return;
  const float* row = xt + (size_t)wave * CT;
  float acc = 0.f;
  for (int c = lane; c < CT; c += 32) acc += row[c] * qw[c];
  #pragma unroll
  for (int off = 16; off >= 1; off >>= 1) acc += __shfl_xor(acc, off, 32);
  if (lane == 0) q[wave] = acc + qb[0];
}

// ---------------------------------------------------------------------------
// 2) in-place softmax over the token axis, one block per batch
// ---------------------------------------------------------------------------
__global__ __launch_bounds__(1024) void softmax_kernel(float* __restrict__ q)
{
  __shared__ float red[32];
  __shared__ float s_max, s_sum;
  float* row = q + (size_t)blockIdx.x * NTOK;
  int tid = threadIdx.x, lane = tid & 31, wid = tid >> 5;

  float m = -INFINITY;
  for (int i = tid; i < NTOK; i += 1024) m = fmaxf(m, row[i]);
  #pragma unroll
  for (int off = 16; off >= 1; off >>= 1) m = fmaxf(m, __shfl_xor(m, off, 32));
  if (lane == 0) red[wid] = m;
  __syncthreads();
  if (wid == 0) {
    float v = red[lane];
    #pragma unroll
    for (int off = 16; off >= 1; off >>= 1) v = fmaxf(v, __shfl_xor(v, off, 32));
    if (lane == 0) s_max = v;
  }
  __syncthreads();
  m = s_max;

  float s = 0.f;
  for (int i = tid; i < NTOK; i += 1024) s += expf(row[i] - m);
  #pragma unroll
  for (int off = 16; off >= 1; off >>= 1) s += __shfl_xor(s, off, 32);
  if (lane == 0) red[wid] = s;
  __syncthreads();
  if (wid == 0) {
    float v = red[lane];
    #pragma unroll
    for (int off = 16; off >= 1; off >>= 1) v += __shfl_xor(v, off, 32);
    if (lane == 0) s_sum = v;
  }
  __syncthreads();
  float inv = 1.f / s_sum;
  for (int i = tid; i < NTOK; i += 1024) row[i] = expf(row[i] - m) * inv;
}

__global__ __launch_bounds__(256) void zero_kernel(float* __restrict__ p, int n)
{
  int i = blockIdx.x * blockDim.x + threadIdx.x;
  for (; i < n; i += gridDim.x * blockDim.x) p[i] = 0.f;
}

// ---------------------------------------------------------------------------
// Fragment loaders (from LDS tiles): 16-bit A-layout, lane holds row
// (lane&15); koff=(lane>=16)?8:0; element e<8 -> K=koff+e, e>=8 -> koff+16+e-8
// LDS tile layout: row-major [64][32] f32, exactly as TDM deposits it.
// ---------------------------------------------------------------------------
__device__ __forceinline__ v16h load_frag(const float* __restrict__ p)
{
  v16h f;
  #pragma unroll
  for (int i = 0; i < 8; i++) {
    f[i]     = (_Float16)p[i];
    f[i + 8] = (_Float16)p[i + 16];
  }
  return f;
}

__device__ __forceinline__ v16h load_frag_scaled(const float* __restrict__ p,
                                                 const float* __restrict__ sc)
{
  v16h f;
  #pragma unroll
  for (int i = 0; i < 8; i++) {
    f[i]     = (_Float16)(p[i]      * sc[i]);
    f[i + 8] = (_Float16)(p[i + 16] * sc[i + 16]);
  }
  return f;
}

// ---------------------------------------------------------------------------
// 3) kv GEMM (M=16384, N=2C, K=C) fused with ctx reduction / relu(v) store.
//    4 waves per block; 64x64 block tile; A/B K-slices staged via TDM.
// ---------------------------------------------------------------------------
__global__ __launch_bounds__(128) void kv_ctx_kernel(
    const float* __restrict__ x,      // [MTOT, C]
    const float* __restrict__ w,      // [2C, C] row-major (d, c)
    const float* __restrict__ bias,   // [2C]
    const float* __restrict__ scores, // [MTOT]
    float* __restrict__ ctx,          // [BATCH, C]  (atomic accum)
    float* __restrict__ reluv,        // [MTOT, C]
    int C)
{
  __shared__ float A_s[64 * 32];
  __shared__ float B_s[64 * 32];
  int lane = threadIdx.x & 31;
  int wid  = threadIdx.x >> 5;            // 0..3
  int wm = wid >> 1, wn = wid & 1;        // 2x2 wave grid over 64x64 tile
  int tm = blockIdx.x, tn = blockIdx.y;

  uint32_t lds_a = (uint32_t)(uintptr_t)(void*)A_s;
  uint32_t lds_b = (uint32_t)(uintptr_t)(void*)B_s;
  int row_a_l = wm * 32 + (lane & 15);    // local A row (i=0 sub-tile)
  int row_b_l = wn * 32 + (lane & 15);    // local B row (j=0 sub-tile)
  int koff    = (lane >> 4) * 8;

  v8f acc[2][2] = {};
  for (int kb = 0; kb < C; kb += 32) {
    if (wid == 0) {
      uint64_t ga = (uint64_t)(uintptr_t)x +
                    4ull * ((uint64_t)(tm * 64) * (uint32_t)C + (uint32_t)kb);
      uint64_t gb = (uint64_t)(uintptr_t)w +
                    4ull * ((uint64_t)(tn * 64) * (uint32_t)C + (uint32_t)kb);
      tdm_load_2d(ga, lds_a, (uint32_t)C, (uint32_t)MTOT,  (uint32_t)C, 32, 64);
      tdm_load_2d(gb, lds_b, (uint32_t)C, (uint32_t)(2*C), (uint32_t)C, 32, 64);
      __builtin_amdgcn_s_wait_tensorcnt(0);
    }
    __syncthreads();
    const float* pa0 = &A_s[row_a_l * 32 + koff];
    const float* pa1 = pa0 + 16 * 32;
    const float* pb0 = &B_s[row_b_l * 32 + koff];
    const float* pb1 = pb0 + 16 * 32;
    v16h a0 = load_frag(pa0), a1 = load_frag(pa1);
    v16h b0 = load_frag(pb0), b1 = load_frag(pb1);
    acc[0][0] = wmma_f16(a0, b0, acc[0][0]);
    acc[0][1] = wmma_f16(a0, b1, acc[0][1]);
    acc[1][0] = wmma_f16(a1, b0, acc[1][0]);
    acc[1][1] = wmma_f16(a1, b1, acc[1][1]);
    __syncthreads();
  }

  int nloc  = lane & 15;
  int mhalf = (lane >> 4) * 8;
  int dbase = tn * 64 + wn * 32;
  bool is_k = (dbase < C);                // uniform per wave (C % 32 == 0)
  int batch = (tm * 64) >> 12;            // 4096 tokens per batch

  #pragma unroll
  for (int j = 0; j < 2; j++) {
    int d = dbase + j * 16 + nloc;
    float bv = bias[d];
    if (is_k) {
      float partial = 0.f;
      #pragma unroll
      for (int i = 0; i < 2; i++) {
        int mbase = tm * 64 + wm * 32 + i * 16 + mhalf;
        #pragma unroll
        for (int r = 0; r < 8; r++)
          partial += (acc[i][j][r] + bv) * scores[mbase + r];
      }
      partial += __shfl_xor(partial, 16, 32);   // combine the two M-halves
      if (lane < 16) atomicAdd(&ctx[batch * C + d], partial);
    } else {
      int c = d - C;
      #pragma unroll
      for (int i = 0; i < 2; i++) {
        int mbase = tm * 64 + wm * 32 + i * 16 + mhalf;
        #pragma unroll
        for (int r = 0; r < 8; r++)
          reluv[(size_t)(mbase + r) * C + c] = fmaxf(acc[i][j][r] + bv, 0.f);
      }
    }
  }
}

// ---------------------------------------------------------------------------
// 4) out GEMM: out[t,d] = sum_c (relu_v[t,c]*ctx[b,c]) * out_w[d,c] + out_b[d]
// ---------------------------------------------------------------------------
__global__ __launch_bounds__(128) void out_kernel(
    const float* __restrict__ reluv,  // [MTOT, C]
    const float* __restrict__ ctx,    // [BATCH, C]
    const float* __restrict__ w,      // [C, C]
    const float* __restrict__ bias,   // [C]
    float* __restrict__ out,          // [MTOT, C]
    int C)
{
  __shared__ float A_s[64 * 32];
  __shared__ float B_s[64 * 32];
  __shared__ float s_ctx[512];
  int lane = threadIdx.x & 31;
  int wid  = threadIdx.x >> 5;
  int wm = wid >> 1, wn = wid & 1;
  int tm = blockIdx.x, tn = blockIdx.y;
  int batch = (tm * 64) >> 12;
  for (int c = threadIdx.x; c < C; c += 128) s_ctx[c] = ctx[batch * C + c];

  uint32_t lds_a = (uint32_t)(uintptr_t)(void*)A_s;
  uint32_t lds_b = (uint32_t)(uintptr_t)(void*)B_s;
  int row_a_l = wm * 32 + (lane & 15);
  int row_b_l = wn * 32 + (lane & 15);
  int koff    = (lane >> 4) * 8;

  v8f acc[2][2] = {};
  for (int kb = 0; kb < C; kb += 32) {
    if (wid == 0) {
      uint64_t ga = (uint64_t)(uintptr_t)reluv +
                    4ull * ((uint64_t)(tm * 64) * (uint32_t)C + (uint32_t)kb);
      uint64_t gb = (uint64_t)(uintptr_t)w +
                    4ull * ((uint64_t)(tn * 64) * (uint32_t)C + (uint32_t)kb);
      tdm_load_2d(ga, lds_a, (uint32_t)C, (uint32_t)MTOT, (uint32_t)C, 32, 64);
      tdm_load_2d(gb, lds_b, (uint32_t)C, (uint32_t)C,    (uint32_t)C, 32, 64);
      __builtin_amdgcn_s_wait_tensorcnt(0);
    }
    __syncthreads();
    const float* pa0 = &A_s[row_a_l * 32 + koff];
    const float* pa1 = pa0 + 16 * 32;
    const float* pb0 = &B_s[row_b_l * 32 + koff];
    const float* pb1 = pb0 + 16 * 32;
    const float* sc  = &s_ctx[kb + koff];
    v16h a0 = load_frag_scaled(pa0, sc), a1 = load_frag_scaled(pa1, sc);
    v16h b0 = load_frag(pb0),            b1 = load_frag(pb1);
    acc[0][0] = wmma_f16(a0, b0, acc[0][0]);
    acc[0][1] = wmma_f16(a0, b1, acc[0][1]);
    acc[1][0] = wmma_f16(a1, b0, acc[1][0]);
    acc[1][1] = wmma_f16(a1, b1, acc[1][1]);
    __syncthreads();
  }

  int nloc  = lane & 15;
  int mhalf = (lane >> 4) * 8;
  #pragma unroll
  for (int j = 0; j < 2; j++) {
    int d = tn * 64 + wn * 32 + j * 16 + nloc;
    float bv = bias[d];
    #pragma unroll
    for (int i = 0; i < 2; i++) {
      int mbase = tm * 64 + wm * 32 + i * 16 + mhalf;
      #pragma unroll
      for (int r = 0; r < 8; r++)
        out[(size_t)(mbase + r) * C + d] = acc[i][j][r] + bv;
    }
  }
}

// ---------------------------------------------------------------------------
extern "C" void kernel_launch(void* const* d_in, const int* in_sizes, int n_in,
                              void* d_out, int out_size, void* d_ws, size_t ws_size,
                              hipStream_t stream)
{
  (void)in_sizes; (void)n_in; (void)out_size; (void)ws_size;
  const float* x[4]  = {(const float*)d_in[0], (const float*)d_in[1],
                        (const float*)d_in[2], (const float*)d_in[3]};
  const float* xt    = (const float*)d_in[4];
  const float* qw    = (const float*)d_in[5];
  const float* qb    = (const float*)d_in[6];

  float* ws     = (float*)d_ws;
  float* scores = ws;              // 16384 floats (q -> softmax in place)
  float* ctxbuf = ws + 16384;      // 4*(64+128+256+512) = 3840 floats
  float* reluv  = ws + 32768;      // up to 16384*512 floats (reused per branch)

  q_kernel<<<MTOT / 8, 256, 0, stream>>>(xt, qw, qb, scores);
  softmax_kernel<<<BATCH, 1024, 0, stream>>>(scores);
  zero_kernel<<<4, 256, 0, stream>>>(ctxbuf, BATCH * CT);

  const int Cs[4] = {64, 128, 256, 512};
  size_t ctx_off = 0, out_off = 0;
  for (int i = 0; i < 4; i++) {
    int C = Cs[i];
    const float* kvw = (const float*)d_in[7 + 4 * i];
    const float* kvb = (const float*)d_in[8 + 4 * i];
    const float* ow  = (const float*)d_in[9 + 4 * i];
    const float* ob  = (const float*)d_in[10 + 4 * i];

    dim3 gkv(MTOT / 64, (2 * C) / 64);
    kv_ctx_kernel<<<gkv, 128, 0, stream>>>(x[i], kvw, kvb, scores,
                                           ctxbuf + ctx_off, reluv, C);
    dim3 go(MTOT / 64, C / 64);
    out_kernel<<<go, 128, 0, stream>>>(reluv, ctxbuf + ctx_off, ow, ob,
                                       (float*)d_out + out_off, C);
    ctx_off += (size_t)BATCH * C;
    out_off += (size_t)MTOT * C;
  }
}
